// MultiHeadAttention_80951543595913
// MI455X (gfx1250) — matrix operand order repaired and checked by reading, not compiled, not run on hardware
//
#include <hip/hip_runtime.h>
#include <hip/hip_bf16.h>

// ---------------------------------------------------------------------------
// Fastformer additive attention for MI455X (gfx1250, wave32, WMMA).
// All four big 16384x1024x1024 GEMMs run on v_wmma_f32_16x16x32_bf16, with
// double-buffered LDS tiles streamed by GLOBAL_LOAD_ASYNC_TO_LDS_B128
// (ASYNCcnt-tracked) when the toolchain exposes the builtin.
// ---------------------------------------------------------------------------

#define NB   4
#define TT   4096
#define DM   1024
#define HH   16
#define DKK  64
#define MROWS (NB * TT)          // 16384 rows for the big GEMMs

typedef __attribute__((ext_vector_type(16))) __bf16 v16bf;
typedef __attribute__((ext_vector_type(8)))  float  v8f;

union AFrag {                    // 16 bf16 = 32 bytes = two b128 LDS loads
    uint4 q[2];
    v16bf v;
};

#if __has_builtin(__builtin_amdgcn_global_load_async_to_lds_b128) && \
    __has_builtin(__builtin_amdgcn_s_wait_asynccnt)
#define USE_ASYNC_LDS 1
typedef int v4i_t __attribute__((vector_size(16)));
typedef __attribute__((address_space(1))) v4i_t* gas1_v4i;   // global int4*
typedef __attribute__((address_space(3))) v4i_t* las3_v4i;   // LDS int4*
#else
#define USE_ASYNC_LDS 0
#endif

__device__ __forceinline__ unsigned int f2bf(float f) {
    union { float f; unsigned int u; } c; c.f = f;
    unsigned int u = c.u;
    u += 0x7FFFu + ((u >> 16) & 1u);       // round-to-nearest-even
    return u >> 16;
}

// ---------------------------------------------------------------------------
// Prep: activations fp32 -> bf16 (row-major, K-contiguous)
// ---------------------------------------------------------------------------
__global__ __launch_bounds__(256)
void convert_x_kernel(const float* __restrict__ X, unsigned short* __restrict__ Xb) {
    int i4 = (blockIdx.x * 256 + threadIdx.x) * 4;
    float4 d = *(const float4*)&X[i4];
    uint2 pk;
    pk.x = f2bf(d.x) | (f2bf(d.y) << 16);
    pk.y = f2bf(d.z) | (f2bf(d.w) << 16);
    *(uint2*)&Xb[i4] = pk;
}

// ---------------------------------------------------------------------------
// Prep: fp32 W[K][N] -> bf16 Wt[N][K]  (K-contiguous for B fragments)
// ---------------------------------------------------------------------------
__global__ void convert_wt_kernel(const float* __restrict__ W,
                                  unsigned short* __restrict__ Wt,
                                  int K, int Nn) {
    int idx = blockIdx.x * 256 + threadIdx.x;
    if (idx >= K * Nn) return;
    int k  = idx / Nn;
    int nn = idx - k * Nn;
    Wt[(size_t)nn * K + k] = (unsigned short)f2bf(W[(size_t)k * Nn + nn]);
}

// ---------------------------------------------------------------------------
// WMMA GEMM: C[M][N] = A[M][K] * Wt[N][K]^T + bias   (A, Wt both bf16)
//   Block tile 128(M) x 64(N), k-step 32, 8 waves of 32, each wave 32x32.
//   Double-buffered LDS; tiles streamed with async global->LDS b128 copies.
//   SCATTER: store fp32 to [n][h][t][d] layout (projections), else row-major.
// gfx1250 16-bit WMMA operand layout (ISA 7.12.2):
//   A frag, lane L: m=L&15, kh=L>>4; VGPR0-3 = K 8kh..8kh+7, VGPR4-7 = K 16+8kh..
//   B frag, lane L: n=L&15, kh=L>>4; VGPR0-3 = K 8kh..8kh+7, VGPR4-7 = K 16+8kh.. (col n of B)
//   C/D,    lane L: n=L&15; VGPR v holds row v + 8*(L>>4)
// ---------------------------------------------------------------------------
template <bool SCATTER>
__global__ __launch_bounds__(256)
void wmma_gemm_kernel(const unsigned short* __restrict__ A,
                      const unsigned short* __restrict__ Bt,
                      const float* __restrict__ bias,
                      float* __restrict__ out,
                      int Kdim, int Ncols) {
    __shared__ __align__(16) unsigned short ldsA[2][128 * 40];  // pad 40: conflict-free b128
    __shared__ __align__(16) unsigned short ldsB[2][64 * 40];

    const int tid     = threadIdx.x;
    const int rowBase = blockIdx.x * 128;
    const int colBase = blockIdx.y * 64;
    const int wid     = tid >> 5;
    const int lane    = tid & 31;
    const int waveM   = wid >> 1;        // 0..3 -> 32-row strip
    const int waveN   = wid & 1;         // 0..1 -> 32-col strip
    const int lm      = lane & 15;
    const int kh      = lane >> 4;
    const int nk      = Kdim >> 5;

    // stage one 128x32 A tile + 64x32 B tile into LDS buffer `buf`
    auto stage = [&](int ks, int buf) {
        const int k0 = ks * 32;
#if USE_ASYNC_LDS
        #pragma unroll
        for (int i = 0; i < 2; ++i) {                    // A: 512 b128 chunks
            int lin = tid + 256 * i;
            int r = lin >> 2, ch = lin & 3;
            const unsigned short* gp = &A[(size_t)(rowBase + r) * Kdim + k0 + ch * 8];
            unsigned short* lp = &ldsA[buf][r * 40 + ch * 8];
            __builtin_amdgcn_global_load_async_to_lds_b128(
                (gas1_v4i)(void*)(const void*)gp, (las3_v4i)(void*)lp, 0, 0);
        }
        {                                                // B: 256 b128 chunks
            int r = tid >> 2, ch = tid & 3;
            const unsigned short* gp = &Bt[(size_t)(colBase + r) * Kdim + k0 + ch * 8];
            unsigned short* lp = &ldsB[buf][r * 40 + ch * 8];
            __builtin_amdgcn_global_load_async_to_lds_b128(
                (gas1_v4i)(void*)(const void*)gp, (las3_v4i)(void*)lp, 0, 0);
        }
#else
        #pragma unroll
        for (int i = 0; i < 2; ++i) {
            int lin = tid + 256 * i;
            int r = lin >> 2, ch = lin & 3;
            uint4 d = *(const uint4*)&A[(size_t)(rowBase + r) * Kdim + k0 + ch * 8];
            *(uint4*)&ldsA[buf][r * 40 + ch * 8] = d;
        }
        {
            int r = tid >> 2, ch = tid & 3;
            uint4 d = *(const uint4*)&Bt[(size_t)(colBase + r) * Kdim + k0 + ch * 8];
            *(uint4*)&ldsB[buf][r * 40 + ch * 8] = d;
        }
#endif
    };

    v8f acc[2][2];
    #pragma unroll
    for (int j = 0; j < 2; ++j) {
        float b = bias[colBase + waveN * 32 + j * 16 + lm];
        #pragma unroll
        for (int i = 0; i < 2; ++i)
            #pragma unroll
            for (int e = 0; e < 8; ++e)
                acc[i][j][e] = b;        // every C row in this column gets bias[c]
    }

    stage(0, 0);                                   // prologue
    for (int k = 0; k < nk; ++k) {
        const int buf = k & 1;
        if (k + 1 < nk) {
            stage(k + 1, buf ^ 1);                 // overlap next tile with compute
#if USE_ASYNC_LDS
            __builtin_amdgcn_s_wait_asynccnt(3);   // retire older group only
#endif
        } else {
#if USE_ASYNC_LDS
            __builtin_amdgcn_s_wait_asynccnt(0);
#endif
        }
        __syncthreads();                           // group-k data visible to all waves

        AFrag a[2], b[2];
        #pragma unroll
        for (int i = 0; i < 2; ++i) {
            int r = waveM * 32 + i * 16 + lm;
            a[i].q[0] = *(const uint4*)&ldsA[buf][r * 40 + 8 * kh];
            a[i].q[1] = *(const uint4*)&ldsA[buf][r * 40 + 16 + 8 * kh];
        }
        #pragma unroll
        for (int j = 0; j < 2; ++j) {
            int c = waveN * 32 + j * 16 + lm;
            b[j].q[0] = *(const uint4*)&ldsB[buf][c * 40 + 8 * kh];
            b[j].q[1] = *(const uint4*)&ldsB[buf][c * 40 + 16 + 8 * kh];
        }
        #pragma unroll
        for (int i = 0; i < 2; ++i)
            #pragma unroll
            for (int j = 0; j < 2; ++j)
                acc[i][j] = __builtin_amdgcn_wmma_f32_16x16x32_bf16(
                    false, a[i].v, false, b[j].v, (short)0, acc[i][j], false, false);
        __syncthreads();                           // done reading buf before reuse
    }

    // ---- epilogue ----
    #pragma unroll
    for (int i = 0; i < 2; ++i) {
        #pragma unroll
        for (int j = 0; j < 2; ++j) {
            int c = colBase + waveN * 32 + j * 16 + lm;
            #pragma unroll
            for (int e = 0; e < 8; ++e) {
                int r = rowBase + waveM * 32 + i * 16 + e + 8 * kh;
                float v = acc[i][j][e];
                if (SCATTER) {
                    int n = r >> 12, t = r & (TT - 1);
                    int h = c >> 6, d = c & 63;
                    out[((size_t)((n * HH + h) * TT + t)) * DKK + d] = v;
                } else {
                    out[(size_t)r * Ncols + c] = v;
                }
            }
        }
    }
}

// ---------------------------------------------------------------------------
// Pooling: one block per (n,h). alpha-softmax over q -> global_query,
// then beta-softmax over p = gq*k -> global_key (= gq ∘ Σ beta_t k_t).
// ---------------------------------------------------------------------------
__global__ __launch_bounds__(256)
void pool_kernel(const float* __restrict__ qb, const float* __restrict__ kb,
                 const int* __restrict__ mask,
                 const float* __restrict__ alpha_w, const float* __restrict__ beta_w,
                 float* __restrict__ gk_out) {
    __shared__ float s_sc[TT];
    __shared__ float s_w[64];
    __shared__ float s_gq[64];
    __shared__ float s_red[256];
    __shared__ float s_part[256];

    const int tid = threadIdx.x;
    const int nh  = blockIdx.x;
    const int n   = nh >> 4;
    const float scale  = 0.125f;                 // DK^-0.5
    const float NEGBIG = -3.402823466e38f;       // finfo(f32).min

    const float* __restrict__ qs = qb + (size_t)nh * TT * DKK;
    const float* __restrict__ ks = kb + (size_t)nh * TT * DKK;

    // ===================== pass A: alpha over q =====================
    if (tid < 64) s_w[tid] = alpha_w[tid];
    __syncthreads();

    float lmax = NEGBIG;
    for (int t = tid; t < TT; t += 256) {
        const float* row = qs + (size_t)t * DKK;
        float dot = 0.f;
        #pragma unroll
        for (int d = 0; d < DKK; d += 4) {
            float4 x = *(const float4*)(row + d);
            dot += x.x * s_w[d] + x.y * s_w[d + 1] + x.z * s_w[d + 2] + x.w * s_w[d + 3];
        }
        float s = dot * scale;
        if (mask[n * TT + t] == 0) s = NEGBIG;
        s_sc[t] = s;
        lmax = fmaxf(lmax, s);
    }
    s_red[tid] = lmax; __syncthreads();
    for (int st = 128; st > 0; st >>= 1) {
        if (tid < st) s_red[tid] = fmaxf(s_red[tid], s_red[tid + st]);
        __syncthreads();
    }
    const float mxA = s_red[0]; __syncthreads();

    float lsum = 0.f;
    for (int t = tid; t < TT; t += 256) {
        float e = expf(s_sc[t] - mxA);
        s_sc[t] = e;
        lsum += e;
    }
    s_red[tid] = lsum; __syncthreads();
    for (int st = 128; st > 0; st >>= 1) {
        if (tid < st) s_red[tid] += s_red[tid + st];
        __syncthreads();
    }
    const float invA = 1.f / s_red[0];

    {   // weighted sum of q -> global query
        const int d = tid & 63, part = tid >> 6;
        float acc = 0.f;
        const int t0 = part * (TT / 4);
        for (int t = t0; t < t0 + TT / 4; ++t)
            acc += s_sc[t] * qs[(size_t)t * DKK + d];
        s_part[tid] = acc;
    }
    __syncthreads();
    if (tid < 64) {
        float g = (s_part[tid] + s_part[64 + tid] + s_part[128 + tid] + s_part[192 + tid]) * invA;
        s_gq[tid] = g;
        s_w[tid]  = g * beta_w[tid];   // dot(p,beta_w) == dot(k, gq*beta_w)
    }
    __syncthreads();

    // ===================== pass B: beta over p = gq*k =====================
    lmax = NEGBIG;
    for (int t = tid; t < TT; t += 256) {
        const float* row = ks + (size_t)t * DKK;
        float dot = 0.f;
        #pragma unroll
        for (int d = 0; d < DKK; d += 4) {
            float4 x = *(const float4*)(row + d);
            dot += x.x * s_w[d] + x.y * s_w[d + 1] + x.z * s_w[d + 2] + x.w * s_w[d + 3];
        }
        float s = dot * scale;
        if (mask[n * TT + t] == 0) s = NEGBIG;
        s_sc[t] = s;
        lmax = fmaxf(lmax, s);
    }
    s_red[tid] = lmax; __syncthreads();
    for (int st = 128; st > 0; st >>= 1) {
        if (tid < st) s_red[tid] = fmaxf(s_red[tid], s_red[tid + st]);
        __syncthreads();
    }
    const float mxB = s_red[0]; __syncthreads();

    lsum = 0.f;
    for (int t = tid; t < TT; t += 256) {
        float e = expf(s_sc[t] - mxB);
        s_sc[t] = e;
        lsum += e;
    }
    s_red[tid] = lsum; __syncthreads();
    for (int st = 128; st > 0; st >>= 1) {
        if (tid < st) s_red[tid] += s_red[tid + st];
        __syncthreads();
    }
    const float invB = 1.f / s_red[0];

    {   // weighted sum of k, then scale by gq -> global key
        const int d = tid & 63, part = tid >> 6;
        float acc = 0.f;
        const int t0 = part * (TT / 4);
        for (int t = t0; t < t0 + TT / 4; ++t)
            acc += s_sc[t] * ks[(size_t)t * DKK + d];
        s_part[tid] = acc;
    }
    __syncthreads();
    if (tid < 64) {
        float g = (s_part[tid] + s_part[64 + tid] + s_part[128 + tid] + s_part[192 + tid]) * invB;
        gk_out[nh * 64 + tid] = s_gq[tid] * g;
    }
}

// ---------------------------------------------------------------------------
// Fused: u = gk ∘ v ; u_r = u @ Wr + br ; A = u_r + q (bf16, [n][t][h*64]).
// One block per (n, h, 128 t's). K=64 WMMA GEMM against Wr^T staged in LDS.
// ---------------------------------------------------------------------------
__global__ __launch_bounds__(256)
void ur_kernel(const float* __restrict__ vb, const float* __restrict__ qb,
               const float* __restrict__ gk, const unsigned short* __restrict__ Wrt,
               const float* __restrict__ br, unsigned short* __restrict__ Abuf) {
    __shared__ __align__(16) unsigned short ldsU[128 * 72];   // pad 72
    __shared__ __align__(16) unsigned short ldsW[64 * 72];
    __shared__ float s_gk[64];

    const int tid = threadIdx.x;
    const int bid = blockIdx.x;
    const int tb = bid & 31, h = (bid >> 5) & 15, n = bid >> 9;
    const int nh = n * HH + h;
    const int tBase = tb * 128;

    if (tid < 64) s_gk[tid] = gk[nh * 64 + tid];
    #pragma unroll
    for (int i = 0; i < 2; ++i) {              // Wr^T (64x64 bf16) -> LDS
        int lin = tid + 256 * i;
        int r = lin >> 3, ch = lin & 7;
        uint4 d = *(const uint4*)&Wrt[r * 64 + ch * 8];
        *(uint4*)&ldsW[r * 72 + ch * 8] = d;
    }
    __syncthreads();

    #pragma unroll
    for (int i = 0; i < 8; ++i) {              // u tile (128x64) -> LDS bf16
        int lin = tid + 256 * i;
        int r = lin >> 4, c4 = (lin & 15) * 4;
        float4 x = *(const float4*)&vb[((size_t)nh * TT + tBase + r) * DKK + c4];
        x.x *= s_gk[c4];     x.y *= s_gk[c4 + 1];
        x.z *= s_gk[c4 + 2]; x.w *= s_gk[c4 + 3];
        uint2 pk;
        pk.x = f2bf(x.x) | (f2bf(x.y) << 16);
        pk.y = f2bf(x.z) | (f2bf(x.w) << 16);
        *(uint2*)&ldsU[r * 72 + c4] = pk;
    }
    __syncthreads();

    const int wid = tid >> 5, lane = tid & 31;
    const int waveM = wid >> 1, waveN = wid & 1;
    const int lm = lane & 15, kh = lane >> 4;

    v8f acc[2][2];
    #pragma unroll
    for (int j = 0; j < 2; ++j) {
        float b = br[waveN * 32 + j * 16 + lm];
        #pragma unroll
        for (int i = 0; i < 2; ++i)
            #pragma unroll
            for (int e = 0; e < 8; ++e) acc[i][j][e] = b;
    }

    #pragma unroll
    for (int k0 = 0; k0 < 64; k0 += 32) {
        AFrag a[2], b2[2];
        #pragma unroll
        for (int i = 0; i < 2; ++i) {
            int r = waveM * 32 + i * 16 + lm;
            a[i].q[0] = *(const uint4*)&ldsU[r * 72 + k0 + 8 * kh];
            a[i].q[1] = *(const uint4*)&ldsU[r * 72 + k0 + 16 + 8 * kh];
        }
        #pragma unroll
        for (int j = 0; j < 2; ++j) {
            int c = waveN * 32 + j * 16 + lm;
            b2[j].q[0] = *(const uint4*)&ldsW[c * 72 + k0 + 8 * kh];
            b2[j].q[1] = *(const uint4*)&ldsW[c * 72 + k0 + 16 + 8 * kh];
        }
        #pragma unroll
        for (int i = 0; i < 2; ++i)
            #pragma unroll
            for (int j = 0; j < 2; ++j)
                acc[i][j] = __builtin_amdgcn_wmma_f32_16x16x32_bf16(
                    false, a[i].v, false, b2[j].v, (short)0, acc[i][j], false, false);
    }

    #pragma unroll
    for (int i = 0; i < 2; ++i) {
        #pragma unroll
        for (int j = 0; j < 2; ++j) {
            int dcol = waveN * 32 + j * 16 + lm;
            #pragma unroll
            for (int e = 0; e < 8; ++e) {
                int trow = tBase + waveM * 32 + i * 16 + e + 8 * kh;
                float vv = acc[i][j][e] + qb[((size_t)nh * TT + trow) * DKK + dcol];
                Abuf[((size_t)(n * TT + trow)) * DM + h * 64 + dcol] = (unsigned short)f2bf(vv);
            }
        }
    }
}

// ---------------------------------------------------------------------------
extern "C" void kernel_launch(void* const* d_in, const int* in_sizes, int n_in,
                              void* d_out, int out_size, void* d_ws, size_t ws_size,
                              hipStream_t stream) {
    const float* x_k     = (const float*)d_in[0];
    const float* x_v     = (const float*)d_in[1];
    const float* x_q     = (const float*)d_in[2];
    const int*   mask    = (const int*)d_in[3];
    const float* Wk      = (const float*)d_in[4];
    const float* bk      = (const float*)d_in[5];
    const float* Wv      = (const float*)d_in[6];
    const float* bv      = (const float*)d_in[7];
    const float* Wq      = (const float*)d_in[8];
    const float* bq      = (const float*)d_in[9];
    const float* alpha_w = (const float*)d_in[10];
    const float* beta_w  = (const float*)d_in[11];
    const float* Wr      = (const float*)d_in[12];
    const float* br      = (const float*)d_in[13];
    const float* Wfc     = (const float*)d_in[14];
    const float* bfc     = (const float*)d_in[15];
    float* out = (float*)d_out;

    // workspace layout
    char* ws = (char*)d_ws;
    const size_t SZ  = (size_t)MROWS * DM * sizeof(float);           // 64 MB
    const size_t HSZ = SZ / 2;                                       // 32 MB (bf16)
    float*          qbuf  = (float*)(ws);                            // [n][h][t][d] fp32
    float*          kbuf  = (float*)(ws + SZ);
    float*          vbuf  = (float*)(ws + 2 * SZ);
    unsigned short* xqbf  = (unsigned short*)(ws + 3 * SZ);          // bf16 activations
    unsigned short* xkbf  = (unsigned short*)(ws + 3 * SZ + HSZ);
    unsigned short* xvbf  = (unsigned short*)(ws + 3 * SZ + 2 * HSZ);
    unsigned short* Abuf  = xqbf;       // alias: xqbf dead before ur_kernel writes A
    char* wbase = ws + 3 * SZ + 3 * HSZ;
    const size_t WSZ = (size_t)DM * DM * sizeof(unsigned short);     // 2 MB
    unsigned short* Wkt  = (unsigned short*)(wbase);
    unsigned short* Wvt  = (unsigned short*)(wbase + WSZ);
    unsigned short* Wqt  = (unsigned short*)(wbase + 2 * WSZ);
    unsigned short* Wfct = (unsigned short*)(wbase + 3 * WSZ);
    unsigned short* Wrt  = (unsigned short*)(wbase + 4 * WSZ);
    float*          gkbuf = (float*)(wbase + 4 * WSZ + 64 * 64 * sizeof(unsigned short));

    // 1) prep: weights -> bf16 [N][K], activations -> bf16 row-major
    int wblocks = (DM * DM + 255) / 256;
    convert_wt_kernel<<<wblocks, 256, 0, stream>>>(Wk,  Wkt,  DM, DM);
    convert_wt_kernel<<<wblocks, 256, 0, stream>>>(Wv,  Wvt,  DM, DM);
    convert_wt_kernel<<<wblocks, 256, 0, stream>>>(Wq,  Wqt,  DM, DM);
    convert_wt_kernel<<<wblocks, 256, 0, stream>>>(Wfc, Wfct, DM, DM);
    convert_wt_kernel<<<(64 * 64 + 255) / 256, 256, 0, stream>>>(Wr, Wrt, 64, 64);
    int xblocks = (MROWS * DM) / 4 / 256;
    convert_x_kernel<<<xblocks, 256, 0, stream>>>(x_q, xqbf);
    convert_x_kernel<<<xblocks, 256, 0, stream>>>(x_k, xkbf);
    convert_x_kernel<<<xblocks, 256, 0, stream>>>(x_v, xvbf);

    // 2) q/k/v projections (WMMA), scatter fp32 to [n][h][t][d]
    dim3 gg(MROWS / 128, DM / 64);
    wmma_gemm_kernel<true><<<gg, 256, 0, stream>>>(xqbf, Wqt, bq, qbuf, DM, DM);
    wmma_gemm_kernel<true><<<gg, 256, 0, stream>>>(xkbf, Wkt, bk, kbuf, DM, DM);
    wmma_gemm_kernel<true><<<gg, 256, 0, stream>>>(xvbf, Wvt, bv, vbuf, DM, DM);

    // 3) additive-attention pooling -> global key per (n,h)
    pool_kernel<<<NB * HH, 256, 0, stream>>>(qbuf, kbuf, mask, alpha_w, beta_w, gkbuf);

    // 4) fused u = gk∘v, u@Wr + br + q -> bf16 A  (A aliases dead xqbf)
    ur_kernel<<<NB * HH * (TT / 128), 256, 0, stream>>>(vbuf, qbuf, gkbuf, Wrt, br, Abuf);

    // 5) final projection (WMMA, bf16 A) -> fp32 output
    wmma_gemm_kernel<false><<<gg, 256, 0, stream>>>(Abuf, Wfct, bfc, out, DM, DM);
}
